// CapsNet_58729382805831
// MI455X (gfx1250) — matrix operand order repaired and checked by reading, not compiled, not run on hardware
//
#include <hip/hip_runtime.h>
#include <hip/hip_bf16.h>

typedef _Float16 h16;
typedef __attribute__((ext_vector_type(16))) _Float16 v16h;
typedef __attribute__((ext_vector_type(8)))  float    v8f;
typedef unsigned int v4u __attribute__((ext_vector_type(4)));
typedef int          v4i __attribute__((ext_vector_type(4)));
typedef int          v8i __attribute__((ext_vector_type(8)));

#define ROUTES 43264      // 256*13*13
#define KDIM   4096       // 256*4*4
#define NDIM   4096       // 16*256 out channels
#define MREAL  5408       // 32*13*13
#define MPAD   5504       // 43*128

// ---- workspace layout (bytes) ----
#define OFF_H     0u                 // h f16: 32*256*29*29      = 13,778,944 B
#define OFF_W16   13778944u          // pcaps W f16: 4096*4096   = 33,554,432 B
#define OFF_A     47333376u          // im2col f16: 5504*4096    = 45,088,768 B
#define OFF_P     92422144u          // gemm out f16: 5504*4096  = 45,088,768 B
#define OFF_U     137510912u         // u f16: 32*43264*16       = 44,302,336 B
#define OFF_UHAT  181813248u         // u_hat f16: 32*43264*40   = 110,755,840 B
#define OFF_PART  292569088u         // partial s f32: 32*6*40   = 30,720 B
#define OFF_VACC  292599808u         // Vacc f32: 32*40          = 5,120 B

// ---------------- stem conv 3->256, k4 s1, ReLU, store f16 ----------------
__global__ __launch_bounds__(256) void k_stem(const float* __restrict__ x,
                                              const float* __restrict__ w,
                                              const float* __restrict__ bias,
                                              h16* __restrict__ h) {
  int bco = blockIdx.x;            // b*256 + co
  int b = bco >> 8, co = bco & 255;
  __shared__ float ws[48];
  int t = threadIdx.x;
  if (t < 48) ws[t] = w[co * 48 + t];
  __syncthreads();
  int pix = blockIdx.y * 256 + t;
  if (pix >= 841) return;
  int oy = pix / 29, ox = pix % 29;
  float acc = bias[co];
  const float* xb = x + (size_t)b * 3 * 1024;
#pragma unroll
  for (int ci = 0; ci < 3; ++ci)
#pragma unroll
    for (int ky = 0; ky < 4; ++ky)
#pragma unroll
      for (int kx = 0; kx < 4; ++kx)
        acc += xb[ci * 1024 + (oy + ky) * 32 + (ox + kx)] * ws[ci * 16 + ky * 4 + kx];
  acc = fmaxf(acc, 0.f);
  h[((size_t)(b * 256 + co) * 29 + oy) * 29 + ox] = (h16)acc;
}

// ---------------- f32 -> f16 convert (weights) ----------------
__global__ void k_cvt(const float* __restrict__ s, h16* __restrict__ d, int n) {
  int i = blockIdx.x * blockDim.x + threadIdx.x;
  int st = gridDim.x * blockDim.x;
  for (; i < n; i += st) d[i] = (h16)s[i];
}

__global__ void k_zero(float* p, int n) {
  int i = blockIdx.x * blockDim.x + threadIdx.x;
  if (i < n) p[i] = 0.f;
}

// ---------------- im2col (k4 s2, 29->13), pad rows 5408..5503 with 0 ----------------
__global__ __launch_bounds__(256) void k_im2col(const h16* __restrict__ h,
                                                h16* __restrict__ A) {
  int m = blockIdx.x;       // 0..5503
  int c = threadIdx.x;      // input channel 0..255 => k = c*16 + ky*4 + kx
  union { uint4 q[2]; h16 v[16]; } buf;
  if (m < MREAL) {
    int b = m / 169, pix = m % 169;
    int oy = pix / 13, ox = pix % 13;
    const h16* hp = h + ((size_t)(b * 256 + c) * 29 + 2 * oy) * 29 + 2 * ox;
#pragma unroll
    for (int ky = 0; ky < 4; ++ky)
#pragma unroll
      for (int kx = 0; kx < 4; ++kx)
        buf.v[ky * 4 + kx] = hp[ky * 29 + kx];
  } else {
    buf.q[0] = make_uint4(0, 0, 0, 0);
    buf.q[1] = make_uint4(0, 0, 0, 0);
  }
  uint4* dst = (uint4*)(A + (size_t)m * KDIM + c * 16);
  dst[0] = buf.q[0];
  dst[1] = buf.q[1];
}

// ---------------- main GEMM: P[m,n] = A[m,:] . W[n,:] + bias[n] ----------------
// TDM (tensor_load_to_lds) double-buffered staging + WMMA f16 compute.
// LDS tile layout: 128 rows, 40-half (80B) stride = 64B data + 16B pad, produced
// directly by TDM padding (pad_interval=16 dwords, pad_amount=4 dwords).
__global__ __launch_bounds__(256) void k_gemm(const h16* __restrict__ A,
                                              const h16* __restrict__ Bw,
                                              const float* __restrict__ bias,
                                              h16* __restrict__ P) {
  __shared__ __align__(16) h16 As[2][128 * 40];
  __shared__ __align__(16) h16 Bs[2][128 * 40];
  const int t = threadIdx.x;
  const int lane = t & 31, wave = t >> 5;
  const int wm = (wave & 1) * 64;       // 2 waves along M
  const int wn = (wave >> 1) * 32;      // 4 waves along N
  const int r = lane & 15, hi = lane >> 4;
  const int mBase = blockIdx.x * 128, nBase = blockIdx.y * 128;

  v8f acc[4][2];
#pragma unroll
  for (int mi = 0; mi < 4; ++mi)
#pragma unroll
    for (int ni = 0; ni < 2; ++ni)
#pragma unroll
      for (int e = 0; e < 8; ++e) acc[mi][ni][e] = 0.f;

  // ---- TDM descriptor pieces ----
  const unsigned long long gA0 =
      (unsigned long long)(uintptr_t)A + (unsigned long long)mBase * (KDIM * 2);
  const unsigned long long gB0 =
      (unsigned long long)(uintptr_t)Bw + (unsigned long long)nBase * (KDIM * 2);
  const unsigned ldsA =
      (unsigned)(unsigned long long)(__attribute__((address_space(3))) h16*)&As[0][0];
  const unsigned ldsB =
      (unsigned)(unsigned long long)(__attribute__((address_space(3))) h16*)&Bs[0][0];
  // group1: data_size=4B(2), pad_enable, pad_interval=16 dwords(3), pad_amount=4 dwords(3)
  //         tensor_dim0=2048 dwords, tensor_dim1=128, tile_dim0=16 dwords, tile_dim1=128,
  //         tensor_dim0_stride=2048 dwords
  const v8i g1 = { (int)((2u << 16) | (1u << 20) | (3u << 22) | (3u << 25)),
                   0x08000000,   // tensor_dim0 low16 in [31:16]
                   0x00800000,   // tensor_dim1 low16 in [31:16]
                   0x00100000,   // tile_dim0=16 in [31:16]
                   128,          // tile_dim1
                   2048,         // tensor_dim0_stride low32
                   0, 0 };
  const v4i gz4 = { 0, 0, 0, 0 };
  const v8i gz8 = { 0, 0, 0, 0, 0, 0, 0, 0 };

  auto issue = [&](unsigned long long gaddr, unsigned laddr) {
    v4u g0;
    g0.x = 1u;                                   // count=1, user descriptor
    g0.y = laddr;                                // lds_addr
    g0.z = (unsigned)gaddr;                      // global_addr[31:0]
    g0.w = (unsigned)(gaddr >> 32) | 0x80000000u; // global_addr[56:32] | type=2
    __builtin_amdgcn_tensor_load_to_lds(g0, g1, gz4, gz4, gz8, 0);
  };

  if (wave == 0) {       // TDM ignores EXEC; scalar branch => one issue per op
    issue(gA0, ldsA);
    issue(gB0, ldsB);
  }

  const int NK = KDIM / 32;
  for (int kt = 0; kt < NK; ++kt) {
    const int buf = kt & 1;
    __syncthreads();     // everyone done reading buf^1 (chunk kt-1)
    if (wave == 0) {
      if (kt + 1 < NK) {
        issue(gA0 + (unsigned long long)(kt + 1) * 64, ldsA + (buf ^ 1) * 10240u);
        issue(gB0 + (unsigned long long)(kt + 1) * 64, ldsB + (buf ^ 1) * 10240u);
        __builtin_amdgcn_s_wait_tensorcnt(2);  // in-order => chunk kt's pair done
      } else {
        __builtin_amdgcn_s_wait_tensorcnt(0);
      }
    }
    __syncthreads();     // chunk kt tiles visible in LDS
    const h16* Ab = &As[buf][0];
    const h16* Bb = &Bs[buf][0];

    union { uint4 q[2]; v16h v; } af[4], bf[2];
#pragma unroll
    for (int mi = 0; mi < 4; ++mi) {
      // A 16x32 f16 layout: lanes<16 K=0..7,16..23 ; lanes>=16 K=8..15,24..31
      const h16* p = Ab + (wm + mi * 16 + r) * 40 + hi * 8;
      af[mi].q[0] = *(const uint4*)p;
      af[mi].q[1] = *(const uint4*)(p + 16);
    }
#pragma unroll
    for (int ni = 0; ni < 2; ++ni) {
      // B 32x16 f16 layout: lanes<16 K=0..7,16..23 ; lanes>=16 K=8..15,24..31 (column = lane&15)
      const h16* p = Bb + (wn + ni * 16 + r) * 40 + hi * 8;
      bf[ni].q[0] = *(const uint4*)p;
      bf[ni].q[1] = *(const uint4*)(p + 16);
    }
#pragma unroll
    for (int mi = 0; mi < 4; ++mi)
#pragma unroll
      for (int ni = 0; ni < 2; ++ni)
        acc[mi][ni] = __builtin_amdgcn_wmma_f32_16x16x32_f16(
            false, af[mi].v, false, bf[ni].v, (short)0, acc[mi][ni], false, false);
  }

  // epilogue: C/D 16x16 f32 layout: lane = (M>=8?16:0)+N ; VGPR e = M&7
#pragma unroll
  for (int ni = 0; ni < 2; ++ni) {
    int gn = nBase + wn + ni * 16 + r;
    float bv = bias[gn];
#pragma unroll
    for (int mi = 0; mi < 4; ++mi) {
      int gm = mBase + wm + mi * 16 + hi * 8;
#pragma unroll
      for (int e = 0; e < 8; ++e)
        P[(size_t)(gm + e) * NDIM + gn] = (h16)(acc[mi][ni][e] + bv);
    }
  }
}

// ---------------- squash over the 16 capsule components -> u[b][j][16] f16 ----------------
__global__ __launch_bounds__(256) void k_squash_u(const h16* __restrict__ P,
                                                  h16* __restrict__ u) {
  int m = blockIdx.x;   // b*169 + pix
  int t = threadIdx.x;  // r1 (0..255)
  const h16* Pm = P + (size_t)m * NDIM;
  float vals[16], n2 = 0.f;
#pragma unroll
  for (int cap = 0; cap < 16; ++cap) {
    float f = (float)Pm[cap * 256 + t];  // channel = cap*256 + r1
    vals[cap] = f;
    n2 += f * f;
  }
  float scale = (n2 / (1.f + n2)) / sqrtf(n2 + 1e-8f);
  int b = m / 169, pix = m % 169;
  size_t base = ((size_t)b * ROUTES + (size_t)t * 169 + pix) * 16;  // j = r1*169+pix
  union { uint4 q[2]; h16 v[16]; } o;
#pragma unroll
  for (int cap = 0; cap < 16; ++cap) o.v[cap] = (h16)(vals[cap] * scale);
  uint4* dp = (uint4*)(u + base);
  dp[0] = o.q[0];
  dp[1] = o.q[1];
}

// ---------------- u_hat[b][j][c*4+o] = sum_i W[j,c,o,i] * u[b,j,i] ----------------
__global__ __launch_bounds__(256) void k_uhat(const float* __restrict__ rW,
                                              const h16* __restrict__ u,
                                              h16* __restrict__ uhat) {
  int j = blockIdx.x, t = threadIdx.x;
  __shared__ float Wj[640];
  __shared__ float uj[512];
  for (int i = t; i < 640; i += 256) Wj[i] = rW[(size_t)j * 640 + i];
  for (int i = t; i < 512; i += 256) {
    int b = i >> 4, ii = i & 15;
    uj[i] = (float)u[((size_t)b * ROUTES + j) * 16 + ii];
  }
  __syncthreads();
  for (int oi = t; oi < 1280; oi += 256) {
    int b = oi / 40, co = oi % 40;
    float acc = 0.f;
#pragma unroll
    for (int i = 0; i < 16; ++i) acc += Wj[co * 16 + i] * uj[b * 16 + i];
    uhat[((size_t)b * ROUTES + j) * 40 + co] = (h16)acc;
  }
}

// ---------------- one routing pass: logits = u_hat.Vacc, softmax, partial s ----------------
__global__ __launch_bounds__(256) void k_route(const h16* __restrict__ uhat,
                                               const float* __restrict__ Vacc,
                                               float* __restrict__ partial) {
  int b = blockIdx.y, chunk = blockIdx.x, t = threadIdx.x;
  __shared__ float Va[40];
  __shared__ float red[256 * 41];
  if (t < 40) Va[t] = Vacc[b * 40 + t];
  __syncthreads();
  float sloc[40];
#pragma unroll
  for (int k = 0; k < 40; ++k) sloc[k] = 0.f;
  int jend = (chunk + 1) * 8192; if (jend > ROUTES) jend = ROUTES;
  for (int j = chunk * 8192 + t; j < jend; j += 256) {
    union { uint4 q[5]; h16 h[40]; } ub;
    const uint4* up = (const uint4*)(uhat + ((size_t)b * ROUTES + j) * 40);
#pragma unroll
    for (int q = 0; q < 5; ++q) ub.q[q] = up[q];
    float uv[40];
#pragma unroll
    for (int k = 0; k < 40; ++k) uv[k] = (float)ub.h[k];
    float bl[10], mx = -1e30f;
#pragma unroll
    for (int c = 0; c < 10; ++c) {
      float s = 0.f;
#pragma unroll
      for (int o = 0; o < 4; ++o) s += uv[c * 4 + o] * Va[c * 4 + o];
      bl[c] = s;
      mx = fmaxf(mx, s);
    }
    float sum = 0.f;
#pragma unroll
    for (int c = 0; c < 10; ++c) { bl[c] = __expf(bl[c] - mx); sum += bl[c]; }
    float inv = 1.f / sum;
#pragma unroll
    for (int c = 0; c < 10; ++c) {
      float coef = bl[c] * inv;
#pragma unroll
      for (int o = 0; o < 4; ++o) sloc[c * 4 + o] += coef * uv[c * 4 + o];
    }
  }
#pragma unroll
  for (int k = 0; k < 40; ++k) red[t * 41 + k] = sloc[k];
  __syncthreads();
  if (t < 40) {
    float s = 0.f;
    for (int rr = 0; rr < 256; ++rr) s += red[rr * 41 + t];
    partial[((size_t)b * 6 + chunk) * 40 + t] = s;
  }
}

// ---------------- finalize: v = squash(s); Vacc += v; last iter: class softmax ----------------
__global__ __launch_bounds__(320) void k_final(const float* __restrict__ partial,
                                               float* __restrict__ Vacc,
                                               float* __restrict__ out, int last) {
  int t = threadIdx.x;   // 0..319 = (b, c)
  int b = t / 10, c = t % 10;
  float v[4], n2 = 0.f;
#pragma unroll
  for (int o = 0; o < 4; ++o) {
    float s = 0.f;
#pragma unroll
    for (int ch = 0; ch < 6; ++ch) s += partial[((size_t)b * 6 + ch) * 40 + c * 4 + o];
    v[o] = s;
    n2 += s * s;
  }
  float scale = (n2 / (1.f + n2)) / sqrtf(n2 + 1e-8f);
  float len2 = 0.f;
#pragma unroll
  for (int o = 0; o < 4; ++o) {
    v[o] *= scale;
    len2 += v[o] * v[o];
    Vacc[b * 40 + c * 4 + o] += v[o];
  }
  __shared__ float L[320];
  L[t] = sqrtf(len2);
  __syncthreads();
  if (last) {
    float mx = -1e30f;
    for (int k = 0; k < 10; ++k) mx = fmaxf(mx, L[b * 10 + k]);
    float sum = 0.f;
    for (int k = 0; k < 10; ++k) sum += __expf(L[b * 10 + k] - mx);
    out[b * 10 + c] = __expf(L[t] - mx) / sum;
  }
}

extern "C" void kernel_launch(void* const* d_in, const int* in_sizes, int n_in,
                              void* d_out, int out_size, void* d_ws, size_t ws_size,
                              hipStream_t stream) {
  const float* x       = (const float*)d_in[0];
  const float* conv_w  = (const float*)d_in[1];
  const float* conv_b  = (const float*)d_in[2];
  const float* pcaps_w = (const float*)d_in[3];
  const float* pcaps_b = (const float*)d_in[4];
  const float* route_W = (const float*)d_in[5];
  float* out = (float*)d_out;

  char* ws = (char*)d_ws;
  h16*   H    = (h16*)(ws + OFF_H);
  h16*   W16  = (h16*)(ws + OFF_W16);
  h16*   A    = (h16*)(ws + OFF_A);
  h16*   P    = (h16*)(ws + OFF_P);
  h16*   U    = (h16*)(ws + OFF_U);
  h16*   UH   = (h16*)(ws + OFF_UHAT);
  float* PART = (float*)(ws + OFF_PART);
  float* VACC = (float*)(ws + OFF_VACC);

  k_zero<<<5, 256, 0, stream>>>(VACC, 1280);
  k_stem<<<dim3(8192, 4), 256, 0, stream>>>(x, conv_w, conv_b, H);
  k_cvt<<<4096, 256, 0, stream>>>(pcaps_w, W16, NDIM * KDIM);
  k_im2col<<<MPAD, 256, 0, stream>>>(H, A);
  k_gemm<<<dim3(43, 32), 256, 0, stream>>>(A, W16, pcaps_b, P);
  k_squash_u<<<MREAL, 256, 0, stream>>>(P, U);
  k_uhat<<<ROUTES, 256, 0, stream>>>(route_W, U, UH);
  for (int it = 0; it < 8; ++it) {
    k_route<<<dim3(6, 32), 256, 0, stream>>>(UH, VACC, PART);
    k_final<<<1, 320, 0, stream>>>(PART, VACC, out, it == 7 ? 1 : 0);
  }
}
